// StaticSage_78675210928332
// MI455X (gfx1250) — matrix operand order repaired and checked by reading, not compiled, not run on hardware
//
#include <hip/hip_runtime.h>

#define N_NODES 50000
#define N_EDGES 800000
#define IN_DIM  128
#define HID     256
#define B_SEEDS 8192

typedef __attribute__((ext_vector_type(16))) __bf16 bf16x16;
typedef __attribute__((ext_vector_type(8)))  __bf16 bf16x8;
typedef __attribute__((ext_vector_type(8)))  float  f32x8;

// ---------------- utility kernels ----------------

__global__ void k_zero(float* __restrict__ p, unsigned n) {
    unsigned i = blockIdx.x * blockDim.x + threadIdx.x;
    if (i < n) p[i] = 0.0f;
}

__global__ void k_count(const int* __restrict__ dst, float* __restrict__ cnt, int E) {
    int i = blockIdx.x * blockDim.x + threadIdx.x;
    if (i < E) atomicAdd(&cnt[dst[i]], 1.0f);
}

// gather h[src] rows, scatter-add into agg[dst]; one thread per (edge, 4 channels)
__global__ void k_scatter(const float* __restrict__ h, const int* __restrict__ src,
                          const int* __restrict__ dst, float* __restrict__ agg,
                          unsigned total, int cshift) {
    unsigned t = blockIdx.x * blockDim.x + threadIdx.x;
    if (t >= total) return;
    unsigned e  = t >> cshift;
    unsigned c4 = t & ((1u << cshift) - 1u);
    int D = 4 << cshift;
    const float4 v = ((const float4*)(h + (size_t)src[e] * D))[c4];
    float* a = agg + (size_t)dst[e] * D + (size_t)c4 * 4;
    atomicAdd(a + 0, v.x);
    atomicAdd(a + 1, v.y);
    atomicAdd(a + 2, v.z);
    atomicAdd(a + 3, v.w);
}

// neigh-mean + bf16 cast of both GEMM A-operands
__global__ void k_make_frags(const float* __restrict__ agg, const float* __restrict__ cnt,
                             const float* __restrict__ hself, __bf16* __restrict__ An,
                             __bf16* __restrict__ As, unsigned total, int dshift) {
    unsigned i = blockIdx.x * blockDim.x + threadIdx.x;
    if (i >= total) return;
    unsigned row = i >> dshift;
    float c = fmaxf(cnt[row], 1.0f);
    An[i] = (__bf16)(agg[i] / c);
    As[i] = (__bf16)hself[i];
}

// W[K][N] f32 -> Wt[N][K] bf16 (so B fragments are contiguous along K)
__global__ void k_wconv(const float* __restrict__ W, __bf16* __restrict__ Wt,
                        int K, int Nn) {
    int i = blockIdx.x * blockDim.x + threadIdx.x;
    if (i >= K * Nn) return;
    int k = i / Nn, n = i % Nn;
    Wt[(size_t)n * K + k] = (__bf16)W[i];
}

// ---------------- WMMA GEMM: out[M][256] = An@Wn + As@Ws + bias (relu opt) ----------------
// block = 512 threads = 16 waves; blockIdx.x = M/16 tile; wave w owns N-tile w.
template <int K, int RELU>
__global__ __launch_bounds__(512) void k_gemm(const __bf16* __restrict__ An,
                                              const __bf16* __restrict__ As,
                                              const __bf16* __restrict__ Wnt,  // [256][K]
                                              const __bf16* __restrict__ Wst,  // [256][K]
                                              const float*  __restrict__ bias,
                                              float* __restrict__ out) {
    __shared__ __align__(16) __bf16 lA[2][16 * K];
    const unsigned tid  = threadIdx.x;
    const unsigned mblk = blockIdx.x;

    // cooperative stage of both 16xK A tiles into LDS (uint2 = 4 bf16)
    {
        const uint2* gN = (const uint2*)(An + (size_t)mblk * 16 * K);
        const uint2* gS = (const uint2*)(As + (size_t)mblk * 16 * K);
        uint2* sN = (uint2*)lA[0];
        uint2* sS = (uint2*)lA[1];
        const unsigned chunks = 16 * K / 4;
        for (unsigned c = tid; c < chunks; c += 512) { sN[c] = gN[c]; sS[c] = gS[c]; }
    }
    __syncthreads();

    const int lane  = tid & 31;
    const int hf    = lane >> 4;        // lane half selects K sub-range (ISA 7.12.2)
    const int mrow  = lane & 15;
    const int ntile = tid >> 5;
    const int ncol  = lane & 15;
    const int n     = ntile * 16 + ncol;

    f32x8 acc = {};
    #pragma unroll
    for (int k0 = 0; k0 < K; k0 += 32) {
        // A fragment: elems 0..15 -> K = k0 + hf*16 + e (contiguous, matches B)
        const int kbase = k0 + hf * 16;
        bf16x8 nlo = *(const bf16x8*)&lA[0][mrow * K + kbase];
        bf16x8 nhi = *(const bf16x8*)&lA[0][mrow * K + kbase + 8];
        bf16x8 slo = *(const bf16x8*)&lA[1][mrow * K + kbase];
        bf16x8 shi = *(const bf16x8*)&lA[1][mrow * K + kbase + 8];
        bf16x16 aN = __builtin_shufflevector(nlo, nhi, 0,1,2,3,4,5,6,7,8,9,10,11,12,13,14,15);
        bf16x16 aS = __builtin_shufflevector(slo, shi, 0,1,2,3,4,5,6,7,8,9,10,11,12,13,14,15);
        // B fragment: lanes 0-15 K=k0..k0+15, lanes 16-31 K=k0+16..k0+31 (contiguous in Wt)
        const int kb = k0 + hf * 16;
        bf16x16 bN = *(const bf16x16*)&Wnt[(size_t)n * K + kb];
        bf16x16 bS = *(const bf16x16*)&Wst[(size_t)n * K + kb];
        acc = __builtin_amdgcn_wmma_f32_16x16x32_bf16(false, aN, false, bN, (short)0, acc, false, false);
        acc = __builtin_amdgcn_wmma_f32_16x16x32_bf16(false, aS, false, bS, (short)0, acc, false, false);
    }

    const float bv = bias[n];
    #pragma unroll
    for (int v = 0; v < 8; ++v) {            // C/D: VGPR v, row = v + 8*half, col = lane%16
        float val = acc[v] + bv;
        if (RELU) val = fmaxf(val, 0.0f);
        const unsigned m = mblk * 16 + v + 8 * hf;
        out[(size_t)m * HID + n] = val;
    }
}

// ---------------- row L2 normalize (one wave32 per row) ----------------
__global__ void k_l2norm(const float* __restrict__ in, float* __restrict__ out, int rows) {
    int row  = blockIdx.x * (blockDim.x >> 5) + (threadIdx.x >> 5);
    int lane = threadIdx.x & 31;
    if (row >= rows) return;
    const float* p = in + (size_t)row * HID;
    float s = 0.0f;
    #pragma unroll
    for (int c = lane; c < HID; c += 32) { float v = p[c]; s += v * v; }
    #pragma unroll
    for (int off = 16; off > 0; off >>= 1) s += __shfl_down(s, off, 32);
    s = __shfl(s, 0, 32);
    float inv = 1.0f / fmaxf(sqrtf(s), 1e-12f);
    float* q = out + (size_t)row * HID;
    #pragma unroll
    for (int c = lane; c < HID; c += 32) q[c] = p[c] * inv;
}

// ---------------- head: out[b] (+)= dot(g[ids[b]], w) (+ bias on accum pass) ----------------
__global__ void k_head(const float* __restrict__ g, const int* __restrict__ ids,
                       const float* __restrict__ w, const float* __restrict__ bout,
                       float* __restrict__ out, int B, int accum) {
    int b    = blockIdx.x * (blockDim.x >> 5) + (threadIdx.x >> 5);
    int lane = threadIdx.x & 31;
    if (b >= B) return;
    const float* r = g + (size_t)ids[b] * HID;
    float s = 0.0f;
    #pragma unroll
    for (int c = lane; c < HID; c += 32) s += r[c] * w[c];
    #pragma unroll
    for (int off = 16; off > 0; off >>= 1) s += __shfl_down(s, off, 32);
    if (lane == 0) {
        if (accum) out[b] += s + bout[0];
        else       out[b] = s;
    }
}

// ---------------- one full SAGE side ----------------
static void run_side(const float* x, const int* src, const int* dst, const int* ids,
                     const float* W_out_part, const float* b_out, int accum,
                     const __bf16* Wn1t, const __bf16* Ws1t, const float* b1,
                     const __bf16* Wn2t, const __bf16* Ws2t, const float* b2,
                     float* cnt, float* agg, float* h1, __bf16* An, __bf16* As,
                     float* out, hipStream_t stream) {
    const int T = 256;
    k_zero<<<(N_NODES + T - 1) / T, T, 0, stream>>>(cnt, N_NODES);
    k_zero<<<(N_NODES * IN_DIM + T - 1) / T, T, 0, stream>>>(agg, N_NODES * IN_DIM);
    k_count<<<(N_EDGES + T - 1) / T, T, 0, stream>>>(dst, cnt, N_EDGES);

    // layer 1
    const unsigned tot1 = (unsigned)N_EDGES * (IN_DIM / 4);
    k_scatter<<<(tot1 + T - 1) / T, T, 0, stream>>>(x, src, dst, agg, tot1, 5);
    const unsigned e1 = (unsigned)N_NODES * IN_DIM;
    k_make_frags<<<(e1 + T - 1) / T, T, 0, stream>>>(agg, cnt, x, An, As, e1, 7);
    k_gemm<IN_DIM, 1><<<N_NODES / 16, 512, 0, stream>>>(An, As, Wn1t, Ws1t, b1, h1);

    // layer 2
    k_zero<<<(N_NODES * HID + T - 1) / T, T, 0, stream>>>(agg, N_NODES * HID);
    const unsigned tot2 = (unsigned)N_EDGES * (HID / 4);
    k_scatter<<<(tot2 + T - 1) / T, T, 0, stream>>>(h1, src, dst, agg, tot2, 6);
    const unsigned e2 = (unsigned)N_NODES * HID;
    k_make_frags<<<(e2 + T - 1) / T, T, 0, stream>>>(agg, cnt, h1, An, As, e2, 8);
    k_gemm<HID, 0><<<N_NODES / 16, 512, 0, stream>>>(An, As, Wn2t, Ws2t, b2, h1);

    // normalize rows (h1 -> agg), then partial head
    k_l2norm<<<(N_NODES + 7) / 8, 256, 0, stream>>>(h1, agg, N_NODES);
    k_head<<<(B_SEEDS + 7) / 8, 256, 0, stream>>>(agg, ids, W_out_part, b_out, out, B_SEEDS, accum);
}

extern "C" void kernel_launch(void* const* d_in, const int* in_sizes, int n_in,
                              void* d_out, int out_size, void* d_ws, size_t ws_size,
                              hipStream_t stream) {
    const float* x_i   = (const float*)d_in[0];
    const float* x_j   = (const float*)d_in[1];
    const int*   src_i = (const int*)d_in[2];
    const int*   dst_i = (const int*)d_in[3];
    const int*   src_j = (const int*)d_in[4];
    const int*   dst_j = (const int*)d_in[5];
    const int*   ids_i = (const int*)d_in[6];
    const int*   ids_j = (const int*)d_in[7];
    const float* Wn1   = (const float*)d_in[8];
    const float* Ws1   = (const float*)d_in[9];
    const float* b1    = (const float*)d_in[10];
    const float* Wn2   = (const float*)d_in[11];
    const float* Ws2   = (const float*)d_in[12];
    const float* b2    = (const float*)d_in[13];
    const float* W_out = (const float*)d_in[14];
    const float* b_out = (const float*)d_in[15];
    float* out = (float*)d_out;

    char*  base = (char*)d_ws;
    size_t off  = 0;
    auto alloc = [&](size_t bytes) -> void* {
        void* p = base + off;
        off += (bytes + 255) & ~(size_t)255;
        return p;
    };
    __bf16* Wn1t = (__bf16*)alloc((size_t)HID * IN_DIM * 2);
    __bf16* Ws1t = (__bf16*)alloc((size_t)HID * IN_DIM * 2);
    __bf16* Wn2t = (__bf16*)alloc((size_t)HID * HID * 2);
    __bf16* Ws2t = (__bf16*)alloc((size_t)HID * HID * 2);
    float*  cnt  = (float*)alloc((size_t)N_NODES * 4);
    float*  agg  = (float*)alloc((size_t)N_NODES * HID * 4);
    float*  h1   = (float*)alloc((size_t)N_NODES * HID * 4);
    __bf16* An   = (__bf16*)alloc((size_t)N_NODES * HID * 2);
    __bf16* As   = (__bf16*)alloc((size_t)N_NODES * HID * 2);

    const int T = 256;
    k_wconv<<<(IN_DIM * HID + T - 1) / T, T, 0, stream>>>(Wn1, Wn1t, IN_DIM, HID);
    k_wconv<<<(IN_DIM * HID + T - 1) / T, T, 0, stream>>>(Ws1, Ws1t, IN_DIM, HID);
    k_wconv<<<(HID * HID + T - 1) / T, T, 0, stream>>>(Wn2, Wn2t, HID, HID);
    k_wconv<<<(HID * HID + T - 1) / T, T, 0, stream>>>(Ws2, Ws2t, HID, HID);

    run_side(x_i, src_i, dst_i, ids_i, W_out,       b_out, /*accum=*/0,
             Wn1t, Ws1t, b1, Wn2t, Ws2t, b2, cnt, agg, h1, An, As, out, stream);
    run_side(x_j, src_j, dst_j, ids_j, W_out + HID, b_out, /*accum=*/1,
             Wn1t, Ws1t, b1, Wn2t, Ws2t, b2, cnt, agg, h1, An, As, out, stream);
}
